// SAKELayer_20624432955636
// MI455X (gfx1250) — matrix operand order repaired and checked by reading, not compiled, not run on hardware
//
#include <hip/hip_runtime.h>

typedef _Float16 v16h __attribute__((ext_vector_type(16)));
typedef _Float16 v8h  __attribute__((ext_vector_type(8)));
typedef float    v8f  __attribute__((ext_vector_type(8)));

#define BB   2
#define NN   384
#define FD   64
#define HD   64
#define NHD  4
#define CD   32
#define EPSV 1e-5f
#define INFV 1e5f

// fast reciprocal: v_rcp_f32 (~1 ulp), avoids the IEEE div expansion
__device__ __forceinline__ float frcp(float v) { return __builtin_amdgcn_rcpf(v); }
__device__ __forceinline__ float silu_f(float v) { return v * frcp(1.0f + __expf(-v)); }

// B-fragment (32x16 f16): lane (l = L&15 -> column, hi = L>>4), element e -> K = (e&7) + hi*8 + (e&8 ? 16 : 0).
// Pre-swizzled in workspace as [frag][lane][16 halves], so each lane loads 32 contiguous bytes.
__device__ __forceinline__ v16h ldB(const _Float16* __restrict__ wfr, int frag, int lane) {
    return *(const v16h*)(wfr + (((frag << 5) + lane) << 4));
}

// A-fragment (16x32 f16) gathered from row-major f16 LDS (stride HD):
// lane row = L&15; elems 0..7 = K kb+hi*8..+7, elems 8..15 = K kb+16+hi*8..+7 (two b128 loads).
__device__ __forceinline__ v16h ldA(const _Float16* he, int row, int kb, int hi) {
    const v8h* p0 = (const v8h*)(he + row * HD + kb + hi * 8);
    const v8h* p1 = (const v8h*)(he + row * HD + kb + 16 + hi * 8);
    v8h a = *p0, b = *p1;
    return __builtin_shufflevector(a, b, 0,1,2,3,4,5,6,7,8,9,10,11,12,13,14,15);
}

__device__ __forceinline__ v8f wmma16(v16h a, v16h b, v8f c) {
    return __builtin_amdgcn_wmma_f32_16x16x32_f16(false, a, false, b, (short)0, c, false, false);
}

// ---- prep 1: per-node v = h@We0[0:64], w = h@We0[64:128] -----------------------------
__global__ void prep_vw(const float* __restrict__ h, const float* __restrict__ We0,
                        float* __restrict__ vArr, float* __restrict__ wArr) {
    int id = blockIdx.x * 256 + threadIdx.x;
    if (id >= BB * NN * HD) return;
    int nodei = id >> 6, k = id & 63;
    const float* hrow = h + nodei * FD;
    float a = 0.f, b = 0.f;
    for (int f = 0; f < FD; f++) {
        float hv = hrow[f];
        a += hv * We0[f * HD + k];
        b += hv * We0[(FD + f) * HD + k];
    }
    vArr[id] = a;
    wArr[id] = b;
}

// ---- prep 2: swizzle We1 / Wc0 / Wx0 (64x64 f32 row-major) into f16 B-fragments ------
__global__ void prep_swz(const float* __restrict__ We1, const float* __restrict__ Wc0,
                         const float* __restrict__ Wx0, _Float16* __restrict__ wfr) {
    int bid = blockIdx.x;           // 24 blocks: widx(3) x kt(2) x nt(4)
    int widx = bid >> 3, kt = (bid >> 2) & 1, nt = bid & 3;
    const float* W = (widx == 0) ? We1 : (widx == 1) ? Wc0 : Wx0;
    for (int idx = threadIdx.x; idx < 512; idx += 256) {
        int ln = idx >> 4, e = idx & 15;
        int K = kt * 32 + (e & 7) + (ln >> 4) * 8 + ((e & 8) ? 16 : 0);
        int n = nt * 16 + (ln & 15);
        wfr[(((widx * 8 + kt * 4 + nt) << 5) + ln) * 16 + e] = (_Float16)W[K * HD + n];
    }
}

// ---- main: one workgroup (256 thr / 8 waves) per (b,i) node --------------------------
// launch_bounds(256, 4): cap at 256 VGPRs/wave; phase splitting keeps only 8 B-frags live
__global__ __launch_bounds__(256, 4) void sake_main(
    const float* __restrict__ hg,  const float* __restrict__ xg,
    const float* __restrict__ We0, const float* __restrict__ be0,
    const float* __restrict__ be1,
    const float* __restrict__ Ws,  const float* __restrict__ bs,
    const float* __restrict__ bc0,
    const float* __restrict__ Wc1, const float* __restrict__ bc1,
    const float* __restrict__ Wp0, const float* __restrict__ bp0,
    const float* __restrict__ Wp1, const float* __restrict__ bp1,
    const float* __restrict__ bx0, const float* __restrict__ Wx1,
    const float* __restrict__ Wn0, const float* __restrict__ bn0,
    const float* __restrict__ Wn1, const float* __restrict__ bn1,
    const float* __restrict__ lg,
    const float* __restrict__ vArr, const float* __restrict__ wArr,
    const _Float16* __restrict__ wfr,
    float* __restrict__ out)
{
    __shared__ __align__(16) _Float16 he16[NN * HD];     // 48 KB: h_e (f16)
    __shared__ __align__(16) _Float16 semArr[NN * NHD];  // sem logits -> comb weights
    __shared__ float normArr[NN];
    __shared__ float xmxS[8][16][4];                     // per-wave tile x_i - x_j
    __shared__ float gAcc[HD][3];                        // g[n][d] = sum_j xhat_{j,d} * t[j,n]
    __shared__ float vS[HD], c0S[HD], be0S[HD], be1S[HD], bc0S[HD], bx0S[HD], wx1S[HD], hiS[FD];
    __shared__ float WsS[HD * NHD], bsS[NHD];
    __shared__ float xiS[3], xaccS[3], sxhS[3];
    __shared__ float combSum[CD][3], combNorm[CD];
    __shared__ float heagg[NHD * HD];
    __shared__ float t1S[HD], hcombS[HD], n1S[HD];

    const int tid   = threadIdx.x;
    const int node  = blockIdx.x;        // b*N + i
    const int b     = node / NN;
    const int i     = node - b * NN;
    const int nbase = b * NN;

    if (tid < HD) {
        vS[tid]   = vArr[node * HD + tid];
        c0S[tid]  = We0[128 * HD + tid];
        be0S[tid] = be0[tid];
        be1S[tid] = be1[tid];
        bc0S[tid] = bc0[tid];
        bx0S[tid] = bx0[tid];
        wx1S[tid] = Wx1[tid];
        hiS[tid]  = hg[node * FD + tid];
    }
    WsS[tid] = Ws[tid];                  // 256 == HD*NHD
    if (tid < NHD) bsS[tid] = bs[tid];
    if (tid < 3) { xiS[tid] = xg[node * 3 + tid]; xaccS[tid] = 0.f; sxhS[tid] = 0.f; }
    if (tid < HD * 3) ((float*)gAcc)[tid] = 0.f;
    __syncthreads();

    const int lane = tid & 31, w = tid >> 5;
    const int l = lane & 15, hi = lane >> 4;

    float acc_x[3]   = {0.f, 0.f, 0.f};
    float acc_sxh[3] = {0.f, 0.f, 0.f};

    // ============ Phase 1a: norms + h1 A-frags + GEMM1 (h_e = silu(h1@We1+be1)) ============
    for (int jt = w; jt < NN / 16; jt += 8) {
        const int j0 = jt * 16;

        if (hi == 0) {                         // norms + displacements (one row per lane)
            int jj = j0 + l;
            float dx0 = xiS[0] - xg[(nbase + jj) * 3 + 0];
            float dx1 = xiS[1] - xg[(nbase + jj) * 3 + 1];
            float dx2 = xiS[2] - xg[(nbase + jj) * 3 + 2];
            float nr  = __builtin_amdgcn_sqrtf(dx0*dx0 + dx1*dx1 + dx2*dx2 + EPSV);
            normArr[jj] = nr;
            float inv = frcp(nr + EPSV);
            acc_sxh[0] += dx0 * inv; acc_sxh[1] += dx1 * inv; acc_sxh[2] += dx2 * inv;
        }

        // Build h1 A-fragments directly in register layout:
        // h1 = silu(v_i + w_j + norm*c0 + be0)
        const int j = j0 + l;
        const float nr = normArr[j];           // same-wave LDS, in-order
        const float* wrow = wArr + (size_t)(nbase + j) * HD;
        v16h a0, a1;
        #pragma unroll
        for (int e = 0; e < 16; e++) {
            int kk = (e & 7) + hi * 8 + ((e & 8) ? 16 : 0);
            a0[e] = (_Float16)silu_f(vS[kk]      + wrow[kk]      + nr * c0S[kk]      + be0S[kk]);
            a1[e] = (_Float16)silu_f(vS[kk + 32] + wrow[kk + 32] + nr * c0S[kk + 32] + be0S[kk + 32]);
        }

        #pragma unroll
        for (int nt = 0; nt < 4; nt++) {
            v8f acc = {};
            acc = wmma16(a0, ldB(wfr, 0 + nt, lane), acc);
            acc = wmma16(a1, ldB(wfr, 4 + nt, lane), acc);
            int n = nt * 16 + l;
            #pragma unroll
            for (int r = 0; r < 8; r++)
                he16[(j0 + r + hi * 8) * HD + n] = (_Float16)silu_f(acc[r] + be1S[n]);
        }
    }

    // ============ Phase 1b: t = silu(h_e@Wc0+bc0); fold g[n][d] += t * xhat_d ============
    float acc_g[4][3];
    #pragma unroll
    for (int nt = 0; nt < 4; nt++) { acc_g[nt][0] = 0.f; acc_g[nt][1] = 0.f; acc_g[nt][2] = 0.f; }

    for (int jt = w; jt < NN / 16; jt += 8) {
        const int j0 = jt * 16;
        if (hi == 0) {                         // recompute displacements (L2-hot)
            int jj = j0 + l;
            xmxS[w][l][0] = xiS[0] - xg[(nbase + jj) * 3 + 0];
            xmxS[w][l][1] = xiS[1] - xg[(nbase + jj) * 3 + 1];
            xmxS[w][l][2] = xiS[2] - xg[(nbase + jj) * 3 + 2];
        }
        const int j = j0 + l;
        v16h ha0 = ldA(he16, j, 0, hi);        // same-wave rows, LDS in-order
        v16h ha1 = ldA(he16, j, 32, hi);

        float xh[8][3];
        #pragma unroll
        for (int r = 0; r < 8; r++) {
            int rl = r + hi * 8;
            float inv = frcp(normArr[j0 + rl] + EPSV);
            xh[r][0] = xmxS[w][rl][0] * inv;
            xh[r][1] = xmxS[w][rl][1] * inv;
            xh[r][2] = xmxS[w][rl][2] * inv;
        }
        #pragma unroll
        for (int nt = 0; nt < 4; nt++) {
            v8f acc = {};
            acc = wmma16(ha0, ldB(wfr, 8 + nt, lane), acc);
            acc = wmma16(ha1, ldB(wfr, 12 + nt, lane), acc);
            int n = nt * 16 + l;
            #pragma unroll
            for (int r = 0; r < 8; r++) {
                float tv = silu_f(acc[r] + bc0S[n]);
                acc_g[nt][0] += tv * xh[r][0];
                acc_g[nt][1] += tv * xh[r][1];
                acc_g[nt][2] += tv * xh[r][2];
            }
        }
    }

    // ============ Phase 1c: u = silu(h_e@Wx0+bx0); scale_j = u.Wx1; fold x update ============
    for (int jt = w; jt < NN / 16; jt += 8) {
        const int j0 = jt * 16;
        if (hi == 0) {
            int jj = j0 + l;
            xmxS[w][l][0] = xiS[0] - xg[(nbase + jj) * 3 + 0];
            xmxS[w][l][1] = xiS[1] - xg[(nbase + jj) * 3 + 1];
            xmxS[w][l][2] = xiS[2] - xg[(nbase + jj) * 3 + 2];
        }
        const int j = j0 + l;
        v16h ha0 = ldA(he16, j, 0, hi);
        v16h ha1 = ldA(he16, j, 32, hi);

        float accS[8] = {0.f,0.f,0.f,0.f,0.f,0.f,0.f,0.f};
        #pragma unroll
        for (int nt = 0; nt < 4; nt++) {
            v8f acc = {};
            acc = wmma16(ha0, ldB(wfr, 16 + nt, lane), acc);
            acc = wmma16(ha1, ldB(wfr, 20 + nt, lane), acc);
            int n = nt * 16 + l;
            #pragma unroll
            for (int r = 0; r < 8; r++)
                accS[r] += silu_f(acc[r] + bx0S[n]) * wx1S[n];
        }
        #pragma unroll
        for (int m = 1; m < 16; m <<= 1)
            #pragma unroll
            for (int r = 0; r < 8; r++)
                accS[r] += __shfl_xor(accS[r], m, 32);
        if (l == 0) {
            #pragma unroll
            for (int r = 0; r < 8; r++) {
                int rl = r + hi * 8;
                acc_x[0] += accS[r] * xmxS[w][rl][0];
                acc_x[1] += accS[r] * xmxS[w][rl][1];
                acc_x[2] += accS[r] * xmxS[w][rl][2];
            }
        }
    }

    // fold hi-halves / lane partials, then barrier-serialized deterministic accumulation
    #pragma unroll
    for (int nt = 0; nt < 4; nt++)
        #pragma unroll
        for (int d = 0; d < 3; d++)
            acc_g[nt][d] += __shfl_xor(acc_g[nt][d], 16, 32);
    #pragma unroll
    for (int d = 0; d < 3; d++) {
        #pragma unroll
        for (int m = 1; m < 32; m <<= 1) {
            acc_x[d]   += __shfl_xor(acc_x[d],   m, 32);
            acc_sxh[d] += __shfl_xor(acc_sxh[d], m, 32);
        }
    }
    for (int ww = 0; ww < 8; ww++) {
        if (w == ww) {
            if (hi == 0) {
                #pragma unroll
                for (int nt = 0; nt < 4; nt++)
                    #pragma unroll
                    for (int d = 0; d < 3; d++)
                        gAcc[nt * 16 + l][d] += acc_g[nt][d];
            }
            if (lane == 0) {
                #pragma unroll
                for (int d = 0; d < 3; d++) { xaccS[d] += acc_x[d]; sxhS[d] += acc_sxh[d]; }
            }
        }
        __syncthreads();
    }

    // ---------------- Phase 2: attention + aggregation + node MLPs ----------------
    // semantic logits: leaky_relu(h_e @ Ws + bs, 0.2) - INF*eye
    for (int p = tid; p < NN * NHD; p += 256) {
        int jj = p >> 2, hh = p & 3;
        float s = bsS[hh];
        const _Float16* her = he16 + jj * HD;
        for (int k = 0; k < HD; k++) s += (float)her[k] * WsS[k * NHD + hh];
        s = (s > 0.f) ? s : 0.2f * s;
        if (jj == i) s -= INFV;
        semArr[p] = (_Float16)s;
    }
    __syncthreads();

    // per-head softmax(eu)*softmax(sem), renormalized with +EPS; waves 0..3 own heads
    if (w < NHD) {
        const float gam = __expf(lg[w]);
        float mE = -3.0e38f, mS = -3.0e38f;
        for (int jj = lane; jj < NN; jj += 32) {
            float le = -(normArr[jj] + ((jj == i) ? INFV : 0.f)) * gam;
            mE = fmaxf(mE, le);
            mS = fmaxf(mS, (float)semArr[jj * NHD + w]);
        }
        #pragma unroll
        for (int m = 1; m < 32; m <<= 1) {
            mE = fmaxf(mE, __shfl_xor(mE, m, 32));
            mS = fmaxf(mS, __shfl_xor(mS, m, 32));
        }
        float sE = 0.f, sS = 0.f;
        for (int jj = lane; jj < NN; jj += 32) {
            float le = -(normArr[jj] + ((jj == i) ? INFV : 0.f)) * gam;
            sE += __expf(le - mE);
            sS += __expf((float)semArr[jj * NHD + w] - mS);
        }
        #pragma unroll
        for (int m = 1; m < 32; m <<= 1) { sE += __shfl_xor(sE, m, 32); sS += __shfl_xor(sS, m, 32); }
        float invE = frcp(sE), invS = frcp(sS);
        float sc = 0.f;
        for (int jj = lane; jj < NN; jj += 32) {
            float le = -(normArr[jj] + ((jj == i) ? INFV : 0.f)) * gam;
            sc += (__expf(le - mE) * invE) * (__expf((float)semArr[jj * NHD + w] - mS) * invS);
        }
        #pragma unroll
        for (int m = 1; m < 32; m <<= 1) sc += __shfl_xor(sc, m, 32);
        float invC = frcp(sc + EPSV);
        for (int jj = lane; jj < NN; jj += 32) {
            float le = -(normArr[jj] + ((jj == i) ? INFV : 0.f)) * gam;
            float eu = __expf(le - mE) * invE;
            float se = __expf((float)semArr[jj * NHD + w] - mS) * invS;
            semArr[jj * NHD + w] = (_Float16)(eu * se * invC);
        }
    }
    __syncthreads();

    // h_e_agg[h, k] = sum_j comb[j,h] * h_e[j,k]   (256 threads = 4 heads x 64 feats)
    {
        int hh = tid >> 6, k = tid & 63;
        float a = 0.f;
        for (int jj = 0; jj < NN; jj++)
            a += (float)semArr[jj * NHD + hh] * (float)he16[jj * HD + k];
        heagg[hh * HD + k] = a;
    }
    // comb_sum[c][d] = (g[:,d] @ Wc1[:,c] + bc1[c]*sum_j xhat_d) / N ; x_out
    if (tid < CD * 3) {
        int c = tid / 3, d = tid - c * 3;
        float s = 0.f;
        for (int n = 0; n < HD; n++) s += gAcc[n][d] * Wc1[n * CD + c];
        s += bc1[c] * sxhS[d];
        combSum[c][d] = s * (1.f / (float)NN);
    } else if (tid < CD * 3 + 3) {
        int d = tid - CD * 3;
        out[BB * NN * FD + node * 3 + d] = xiS[d] + xaccS[d] * (1.f / (float)NN);
    }
    __syncthreads();
    if (tid < CD)
        combNorm[tid] = combSum[tid][0]*combSum[tid][0] + combSum[tid][1]*combSum[tid][1]
                      + combSum[tid][2]*combSum[tid][2];
    __syncthreads();
    if (tid < HD) {
        float z = bp0[tid];
        for (int c = 0; c < CD; c++) z += combNorm[c] * Wp0[c * HD + tid];
        t1S[tid] = silu_f(z);
    }
    __syncthreads();
    if (tid < HD) {
        float z = bp1[tid];
        for (int m = 0; m < HD; m++) z += t1S[m] * Wp1[m * HD + tid];
        hcombS[tid] = z;
    }
    __syncthreads();
    if (tid < HD) {
        float z = bn0[tid];
        for (int f = 0; f < FD; f++)       z += hiS[f]    * Wn0[f * HD + tid];
        for (int m = 0; m < NHD * HD; m++) z += heagg[m]  * Wn0[(FD + m) * HD + tid];
        for (int m = 0; m < HD; m++)       z += hcombS[m] * Wn0[(FD + NHD * HD + m) * HD + tid];
        n1S[tid] = silu_f(z);
    }
    __syncthreads();
    if (tid < FD) {
        float z = bn1[tid];
        for (int k = 0; k < HD; k++) z += n1S[k] * Wn1[k * FD + tid];
        out[node * FD + tid] = hiS[tid] + z;
    }
}

extern "C" void kernel_launch(void* const* d_in, const int* in_sizes, int n_in,
                              void* d_out, int out_size, void* d_ws, size_t ws_size,
                              hipStream_t stream) {
    (void)in_sizes; (void)n_in; (void)out_size; (void)ws_size;
    const float* h   = (const float*)d_in[0];
    const float* x   = (const float*)d_in[1];
    const float* We0 = (const float*)d_in[2];
    const float* be0 = (const float*)d_in[3];
    const float* We1 = (const float*)d_in[4];
    const float* be1 = (const float*)d_in[5];
    const float* Ws  = (const float*)d_in[6];
    const float* bs  = (const float*)d_in[7];
    const float* Wc0 = (const float*)d_in[8];
    const float* bc0 = (const float*)d_in[9];
    const float* Wc1 = (const float*)d_in[10];
    const float* bc1 = (const float*)d_in[11];
    const float* Wp0 = (const float*)d_in[12];
    const float* bp0 = (const float*)d_in[13];
    const float* Wp1 = (const float*)d_in[14];
    const float* bp1 = (const float*)d_in[15];
    const float* Wx0 = (const float*)d_in[16];
    const float* bx0 = (const float*)d_in[17];
    const float* Wx1 = (const float*)d_in[18];
    const float* Wn0 = (const float*)d_in[19];
    const float* bn0 = (const float*)d_in[20];
    const float* Wn1 = (const float*)d_in[21];
    const float* bn1 = (const float*)d_in[22];
    const float* lg  = (const float*)d_in[23];
    float* out = (float*)d_out;

    float* vArr = (float*)d_ws;                      // 768*64 f32
    float* wArr = vArr + BB * NN * HD;               // 768*64 f32
    _Float16* wfr = (_Float16*)(wArr + BB * NN * HD); // 24 frags * 32 lanes * 16 halves

    prep_vw <<<(BB * NN * HD + 255) / 256, 256, 0, stream>>>(h, We0, vArr, wArr);
    prep_swz<<<24, 256, 0, stream>>>(We1, Wc0, Wx0, wfr);
    sake_main<<<BB * NN, 256, 0, stream>>>(h, x, We0, be0, be1, Ws, bs, bc0, Wc1, bc1,
                                           Wp0, bp0, Wp1, bp1, bx0, Wx1, Wn0, bn0,
                                           Wn1, bn1, lg, vArr, wArr, wfr, out);
}
